// MultiScaleHead_51677046505524
// MI455X (gfx1250) — matrix operand order repaired and checked by reading, not compiled, not run on hardware
//
#include <hip/hip_runtime.h>

// ---------------------------------------------------------------------------
// MultiScaleHead forward for MI455X (gfx1250, wave32, WMMA bf16 16x16x32).
//
// GEMMs/attention matmuls: v_wmma_f32_16x16x32_bf16 (bf16 in, f32 accum).
// GEMM B-panels staged to LDS by the Tensor Data Mover (tensor_load_to_lds,
// double-buffered, s_wait_tensorcnt). Softmax/LayerNorm/LSTM stay f32.
//
// Input index map (params pytree flattened alphabetically, JAX-style, after
// the 9 positional inputs):
//  0 hidden_states (4,8,1024,768) f32     1 last_hidden_state (8,1024,768)
//  2 attention_mask (8,1024)              3 p_head (8,16) i32   4 p_tail i32
//  5 p_mask (8,16) f32                    6 s_head (8,64) i32   7 s_tail i32
//  8 s_mask (8,64) f32
//  params: 9..18  attn{bk,bo,bq,bv,ln_b,ln_g,wk,wo,wq,wv}
//          19 ctx_b  20 ctx_cls_b  21 ctx_cls_w  22 ctx_w  23 layer_weights
//          24 mha_ln_b  25 mha_ln_g
//          26..47 para{attn{...},cls_b,cls_w,ln_b,ln_g,
//                      lstm{b_hh_b,b_hh_f,b_ih_b,b_ih_f,w_hh_b,w_hh_f,
//                           w_ih_b,w_ih_f}}
//          48..69 sent{...same...}
// ---------------------------------------------------------------------------

#define DEV __device__ __forceinline__

typedef __attribute__((ext_vector_type(16))) __bf16 v16bf;
typedef __attribute__((ext_vector_type(8)))  float  v8f;
typedef unsigned int u32x4 __attribute__((ext_vector_type(4)));
typedef int          i32x8 __attribute__((ext_vector_type(8)));
typedef int          i32x4 __attribute__((ext_vector_type(4)));

static constexpr int Bc = 8, Lc = 1024, Hc = 768, NHc = 4, Dc = 192;
static constexpr int HDc = 384, Gc = 1536;

union FragAB {
    v16bf v;
    unsigned short u[16];
    unsigned x[8];
};

DEV unsigned short f2bf(float f) {
    union { float f; unsigned u; } cv;
    cv.f = f;
    unsigned u = cv.u;
    u += 0x7fffu + ((u >> 16) & 1u);   // round-to-nearest-even
    return (unsigned short)(u >> 16);
}

DEV v8f zero8() {
    v8f z;
#pragma unroll
    for (int i = 0; i < 8; ++i) z[i] = 0.f;
    return z;
}

// Per-lane k offset for the 16x32 bf16 A fragment (ISA 7.12.2):
// lanes 0-15: V0..V3 -> K 0..7, V4..V7 -> K 16..23 ; lanes 16-31: +8.
DEV int a_kidx(int j, int lane) {
    int kh = (lane >> 4) * 8;
    return (j < 4 ? 2 * j : 16 + 2 * (j - 4)) + kh;
}

DEV v8f wmma_bf16(const FragAB& a, const FragAB& b, v8f c) {
    return __builtin_amdgcn_wmma_f32_16x16x32_bf16(false, a.v, false, b.v,
                                                   (short)0, c, false, false);
}

// ---------------------------------------------------------------------------
// TDM: async 2D tile load (global -> LDS), D# per CDNA5 ISA ch.8.
// group0: count=1, lds_addr, global_addr, type=2("image")
// group1: data_size=2B, tensor_dim0/1, tile_dim0/1, tensor_dim0_stride
// ---------------------------------------------------------------------------
DEV void tdm_load_2d(const unsigned short* gsrc, void* ldst, unsigned tileW,
                     unsigned tileH, unsigned strideElems, unsigned tensorW,
                     unsigned tensorH) {
    unsigned lds = (unsigned)(size_t)ldst;   // LDS aperture: offset in [31:0]
    unsigned long long ga = (unsigned long long)(size_t)gsrc;
    u32x4 g0;
    g0[0] = 1u;                                        // count=1, user mode
    g0[1] = lds;                                       // lds_addr
    g0[2] = (unsigned)ga;                              // global_addr[31:0]
    g0[3] = (unsigned)((ga >> 32) & 0x1FFFFFFu) | (2u << 30);  // ga[56:32]|type=2
    i32x8 g1;
    g1[0] = (int)(1u << 16);                           // data_size = 2 bytes
    g1[1] = (int)((tensorW & 0xFFFFu) << 16);          // tensor_dim0[15:0]
    g1[2] = (int)((tensorW >> 16) | ((tensorH & 0xFFFFu) << 16));
    g1[3] = (int)(((tensorH >> 16) & 0xFFFFu) | (tileW << 16));  // tile_dim0
    g1[4] = (int)(tileH & 0xFFFFu);                    // tile_dim1, tile_dim2=0
    g1[5] = (int)strideElems;                          // tensor_dim0_stride
    g1[6] = 0;
    g1[7] = 0;
    i32x4 gz;
    gz[0] = gz[1] = gz[2] = gz[3] = 0;
#if __clang_major__ >= 23
    i32x8 z8;
#pragma unroll
    for (int i = 0; i < 8; ++i) z8[i] = 0;
    __builtin_amdgcn_tensor_load_to_lds(g0, g1, gz, gz, z8, 0);
#else
    __builtin_amdgcn_tensor_load_to_lds(g0, g1, gz, gz, 0);
#endif
}

DEV float blk_sum256(float v) {
    __shared__ float red[256];
    int t = threadIdx.x;
    red[t] = v;
    __syncthreads();
    for (int s = 128; s > 0; s >>= 1) {
        if (t < s) red[t] += red[t + s];
        __syncthreads();
    }
    float r = red[0];
    __syncthreads();
    return r;
}

// ---------------------------------------------------------------------------
// Weight prep: W (N,K) f32 row-major -> Wt (K,N) bf16 (B-fragment friendly)
// ---------------------------------------------------------------------------
__global__ void transpose_w_kernel(const float* __restrict__ W,
                                   unsigned short* __restrict__ Wt,
                                   int N, int K) {
    size_t i = (size_t)blockIdx.x * 256 + threadIdx.x;
    if (i >= (size_t)N * K) return;
    int k = (int)(i % K), n = (int)(i / K);
    Wt[(size_t)k * N + n] = f2bf(W[i]);
}

__global__ void to_bf16_kernel(const float* __restrict__ in,
                               unsigned short* __restrict__ out, size_t n) {
    size_t i = (size_t)blockIdx.x * 256 + threadIdx.x;
    if (i < n) out[i] = f2bf(in[i]);
}

__global__ void bias_sum_kernel(const float* __restrict__ a,
                                const float* __restrict__ b,
                                float* __restrict__ o, int n) {
    int i = blockIdx.x * 256 + threadIdx.x;
    if (i < n) o[i] = a[i] + b[i];
}

// ---------------------------------------------------------------------------
// enc0 = LayerNorm(sum_l softmax(lw)_l * hidden_states[l]), eps 1e-7
// ---------------------------------------------------------------------------
__global__ void layer_mix_ln_kernel(const float* __restrict__ hs,
                                    const float* __restrict__ lw,
                                    const float* __restrict__ g,
                                    const float* __restrict__ bt,
                                    float* __restrict__ outf,
                                    unsigned short* __restrict__ outb) {
    int row = blockIdx.x, t = threadIdx.x;
    float w[4], wm = -3.4e38f, ws = 0.f;
    for (int l = 0; l < 4; ++l) wm = fmaxf(wm, lw[l]);
    for (int l = 0; l < 4; ++l) { w[l] = __expf(lw[l] - wm); ws += w[l]; }
    for (int l = 0; l < 4; ++l) w[l] /= ws;

    const size_t lstride = (size_t)Bc * Lc * Hc;
    size_t roff = (size_t)row * Hc;
    float x[3];
#pragma unroll
    for (int i = 0; i < 3; ++i) {
        int c = t + i * 256;
        float s = 0.f;
        for (int l = 0; l < 4; ++l) s += w[l] * hs[l * lstride + roff + c];
        x[i] = s;
    }
    float mean = blk_sum256(x[0] + x[1] + x[2]) * (1.f / Hc);
    float d0 = x[0] - mean, d1 = x[1] - mean, d2 = x[2] - mean;
    float var = blk_sum256(d0 * d0 + d1 * d1 + d2 * d2) * (1.f / Hc);
    float inv = rsqrtf(var + 1e-7f);
#pragma unroll
    for (int i = 0; i < 3; ++i) {
        int c = t + i * 256;
        float y = (x[i] - mean) * inv * g[c] + bt[c];
        outf[roff + c] = y;
        outb[roff + c] = f2bf(y);
    }
}

// out = LayerNorm(x [+ res]) ; optional bf16 copy
__global__ void add_ln_kernel(const float* __restrict__ x,
                              const float* __restrict__ res,
                              const float* __restrict__ g,
                              const float* __restrict__ bt, float eps,
                              float* __restrict__ outf,
                              unsigned short* __restrict__ outb) {
    int row = blockIdx.x, t = threadIdx.x;
    size_t roff = (size_t)row * Hc;
    float v[3];
#pragma unroll
    for (int i = 0; i < 3; ++i) {
        int c = t + i * 256;
        float s = x[roff + c];
        if (res) s += res[roff + c];
        v[i] = s;
    }
    float mean = blk_sum256(v[0] + v[1] + v[2]) * (1.f / Hc);
    float d0 = v[0] - mean, d1 = v[1] - mean, d2 = v[2] - mean;
    float var = blk_sum256(d0 * d0 + d1 * d1 + d2 * d2) * (1.f / Hc);
    float inv = rsqrtf(var + eps);
#pragma unroll
    for (int i = 0; i < 3; ++i) {
        int c = t + i * 256;
        float y = (v[i] - mean) * inv * g[c] + bt[c];
        outf[roff + c] = y;
        if (outb) outb[roff + c] = f2bf(y);
    }
}

// ---------------------------------------------------------------------------
// GEMM: C(M,N) = A(M,K)bf16 @ Bt(K,N)bf16 + bias.
// 4 waves / block, 64x64 tile. Wave w: rows 16w..16w+15, 4 accumulators over
// the 64-wide N strip (A fragment reused 4x). B panel (32x64 bf16, 4 KB) is
// TDM-loaded into LDS, double-buffered on TENSORcnt.
// ---------------------------------------------------------------------------
__global__ __launch_bounds__(128) void gemm_bf16_kernel(
    const unsigned short* __restrict__ A, const unsigned short* __restrict__ Bt,
    const float* __restrict__ bias, int M, int N, int K,
    float* __restrict__ outf, unsigned short* __restrict__ outb) {
    __shared__ unsigned short panel[2][32 * 64];
    int tid = threadIdx.x;
    int lane = tid & 31, wv = tid >> 5;
    int n0 = blockIdx.x * 64;
    int m0 = blockIdx.y * 64 + wv * 16;
    const unsigned short* rowA = A + (size_t)(m0 + (lane & 15)) * K;

    v8f acc[4];
#pragma unroll
    for (int s = 0; s < 4; ++s) acc[s] = zero8();

    int T = K / 32;
    if (wv == 0)
        tdm_load_2d(Bt + (size_t)0 * N + n0, panel[0], 64, 32, (unsigned)N,
                    (unsigned)N, (unsigned)K);
    for (int t = 0; t < T; ++t) {
        if (wv == 0) {
            if (t + 1 < T) {
                tdm_load_2d(Bt + (size_t)(t + 1) * 32 * N + n0,
                            panel[(t + 1) & 1], 64, 32, (unsigned)N,
                            (unsigned)N, (unsigned)K);
                __builtin_amdgcn_s_wait_tensorcnt(1);  // oldest (current) done
            } else {
                __builtin_amdgcn_s_wait_tensorcnt(0);
            }
        }
        __syncthreads();
        const unsigned short* P = panel[t & 1];
        FragAB a;
#pragma unroll
        for (int j = 0; j < 8; ++j)
            a.x[j] = *(const unsigned*)(rowA + t * 32 + a_kidx(j, lane));
#pragma unroll
        for (int sub = 0; sub < 4; ++sub) {
            FragAB b;
#pragma unroll
            for (int j = 0; j < 8; ++j) {
                int kb = a_kidx(j, lane);
                b.u[2 * j]     = P[kb * 64 + sub * 16 + (lane & 15)];
                b.u[2 * j + 1] = P[(kb + 1) * 64 + sub * 16 + (lane & 15)];
            }
            acc[sub] = wmma_bf16(a, b, acc[sub]);
        }
        __syncthreads();   // all waves done with panel before next TDM write
    }

    int mb = m0 + 8 * (lane >> 4);
#pragma unroll
    for (int sub = 0; sub < 4; ++sub) {
        int n = n0 + sub * 16 + (lane & 15);
        float bv = bias ? bias[n] : 0.f;
#pragma unroll
        for (int r = 0; r < 8; ++r) {
            float v = acc[sub][r] + bv;
            size_t idx = (size_t)(mb + r) * N + n;
            if (outf) outf[idx] = v;
            if (outb) outb[idx] = f2bf(v);
        }
    }
}

// Kb (B,L,768) bf16 -> Kt (B,4,192,L) bf16 (so B-fragment rows are contiguous)
__global__ void transpose_k_kernel(const unsigned short* __restrict__ Kb,
                                   unsigned short* __restrict__ Kt, int Lx) {
    size_t total = (size_t)Bc * Lx * Hc;
    size_t i = (size_t)blockIdx.x * 256 + threadIdx.x;
    if (i >= total) return;
    int col = (int)(i % Hc);
    size_t tmp = i / Hc;
    int l = (int)(tmp % Lx), b = (int)(tmp / Lx);
    int h = col / Dc, d = col % Dc;
    Kt[((size_t)(b * NHc + h) * Dc + d) * Lx + l] = Kb[i];
}

// ---------------------------------------------------------------------------
// Attention: per (b,h,16-q stripe): scores = Q.Kt/sqrt(d)+mask -> softmax ->
// ctx = attn @ V. Q fragments preloaded once; ctx phase keeps 12 accumulators
// so each attn A-fragment is built once per 32-k step.
// ---------------------------------------------------------------------------
__global__ void attention_kernel(const unsigned short* __restrict__ Qb,
                                 const unsigned short* __restrict__ Kt,
                                 const unsigned short* __restrict__ Vb,
                                 const float* __restrict__ mask, int Lx,
                                 float scale,
                                 unsigned short* __restrict__ ctxo) {
    extern __shared__ float s[];
    int lane = threadIdx.x;
    int q0 = blockIdx.x * 16, h = blockIdx.y, b = blockIdx.z;
    int Lpad = (Lx + 31) & ~31;

    int tail = Lpad - Lx;                       // zero pad (only L=16 case)
    for (int idx = lane; idx < 16 * tail; idx += 32) {
        int r = idx / tail, c = Lx + idx % tail;
        s[r * Lpad + c] = 0.f;
    }

    int mA = lane & 15;
    const unsigned short* rowQ =
        Qb + ((size_t)(b * Lx + q0 + mA)) * Hc + h * Dc;

    // preload the 6 Q fragments (reused for every K tile)
    FragAB qf[6];
#pragma unroll
    for (int dd = 0; dd < Dc / 32; ++dd)
#pragma unroll
        for (int j = 0; j < 8; ++j)
            qf[dd].x[j] =
                *(const unsigned*)(rowQ + dd * 32 + a_kidx(j, lane));

    // phase 1: scores
    for (int kt = 0; kt < Lx / 16; ++kt) {
        v8f acc = zero8();
#pragma unroll
        for (int dd = 0; dd < Dc / 32; ++dd) {
            int k0 = dd * 32;
            FragAB bm;
#pragma unroll
            for (int j = 0; j < 8; ++j) {
                const unsigned short* kp =
                    Kt + ((size_t)((b * NHc + h) * Dc + k0 + a_kidx(j, lane))) * Lx +
                    kt * 16 + (lane & 15);
                bm.u[2 * j]     = kp[0];
                bm.u[2 * j + 1] = kp[Lx];
            }
            acc = wmma_bf16(qf[dd], bm, acc);
        }
        int n = lane & 15, kcol = kt * 16 + n;
        float madd = (1.f - mask[(size_t)b * Lx + kcol]) * -10000.f;
        int rb = 8 * (lane >> 4);
#pragma unroll
        for (int r = 0; r < 8; ++r)
            s[(rb + r) * Lpad + kcol] = acc[r] * scale + madd;
    }
    __syncthreads();

    // phase 2: row softmax (2 lanes per row, halves joined via shfl_xor 16)
    {
        int r = lane & 15, half = lane >> 4;
        int cn = Lx / 2, c0 = half * cn;
        float mx = -3.4e38f;
        for (int c = c0; c < c0 + cn; ++c) mx = fmaxf(mx, s[r * Lpad + c]);
        mx = fmaxf(mx, __shfl_xor(mx, 16));
        float sum = 0.f;
        for (int c = c0; c < c0 + cn; ++c) {
            float e = __expf(s[r * Lpad + c] - mx);
            s[r * Lpad + c] = e;
            sum += e;
        }
        sum += __shfl_xor(sum, 16);
        float inv = 1.f / sum;
        for (int c = c0; c < c0 + cn; ++c) s[r * Lpad + c] *= inv;
    }
    __syncthreads();

    // phase 3: ctx = attn @ V, 12 accumulators, attn fragment built once/kstep
    v8f acc[12];
#pragma unroll
    for (int dt = 0; dt < 12; ++dt) acc[dt] = zero8();
    for (int kk = 0; kk < Lpad / 32; ++kk) {
        int k0 = kk * 32;
        FragAB a;
#pragma unroll
        for (int j = 0; j < 8; ++j) {
            int ki = k0 + a_kidx(j, lane);
            a.u[2 * j]     = f2bf(s[mA * Lpad + ki]);
            a.u[2 * j + 1] = f2bf(s[mA * Lpad + ki + 1]);
        }
        const unsigned short* rowV =
            Vb + ((size_t)b * Lx) * Hc + h * Dc + (lane & 15);
#pragma unroll
        for (int dt = 0; dt < 12; ++dt) {
            FragAB bm;
#pragma unroll
            for (int j = 0; j < 8; ++j) {
                int kr = k0 + a_kidx(j, lane);
                bm.u[2 * j] = kr < Lx ? rowV[(size_t)kr * Hc + dt * 16] : (unsigned short)0;
                bm.u[2 * j + 1] = kr + 1 < Lx ? rowV[(size_t)(kr + 1) * Hc + dt * 16] : (unsigned short)0;
            }
            acc[dt] = wmma_bf16(a, bm, acc[dt]);   // EXEC uniform here
        }
    }
    {
        int n = lane & 15, rb = 8 * (lane >> 4);
#pragma unroll
        for (int dt = 0; dt < 12; ++dt)
#pragma unroll
            for (int r = 0; r < 8; ++r)
                ctxo[((size_t)(b * Lx + q0 + rb + r)) * Hc + h * Dc + dt * 16 +
                     n] = f2bf(acc[dt][r]);
    }
}

// ---------------------------------------------------------------------------
// Misc scalar kernels
// ---------------------------------------------------------------------------
__global__ void span_mean_kernel(const float* __restrict__ enc,
                                 const int* __restrict__ head,
                                 const int* __restrict__ tail, int N,
                                 float* __restrict__ out) {
    int n = blockIdx.x, b = blockIdx.y, t = threadIdx.x;
    int h0 = head[b * N + n] + 1, t1 = tail[b * N + n];
    float cnt = (float)(t1 - h0);
    float inv = cnt > 0.f ? 1.f / cnt : 0.f;
#pragma unroll
    for (int i = 0; i < 3; ++i) {
        int c = t + i * 256;
        float s = 0.f;
        for (int r = h0; r < t1; ++r) s += enc[((size_t)(b * Lc + r)) * Hc + c];
        out[((size_t)(b * N + n)) * Hc + c] = s * inv;
    }
}

__global__ void mean_pool_kernel(const float* __restrict__ x,
                                 const float* __restrict__ mask, int Lx,
                                 float* __restrict__ out) {
    int b = blockIdx.x, t = threadIdx.x;
    float msum = 0.f;
    for (int l = 0; l < Lx; ++l) msum += mask[(size_t)b * Lx + l];
    float inv = 1.f / fmaxf(msum, 1e-9f);
#pragma unroll
    for (int i = 0; i < 3; ++i) {
        int c = t + i * 256;
        float s = 0.f;
        for (int l = 0; l < Lx; ++l)
            s += x[((size_t)(b * Lx + l)) * Hc + c] * mask[(size_t)b * Lx + l];
        out[(size_t)b * Hc + c] = s * inv;
    }
}

// cls = relu(last_hidden[:,0] @ W^T + b) ; M=8 -> plain GEMV
__global__ void gemv_relu_kernel(const float* __restrict__ lasth,
                                 const float* __restrict__ W,
                                 const float* __restrict__ bias,
                                 float* __restrict__ out) {
    int b = blockIdx.x, t = threadIdx.x;
    const float* x = lasth + (size_t)b * Lc * Hc;   // token 0
#pragma unroll
    for (int i = 0; i < 3; ++i) {
        int n = t + i * 256;
        float s = 0.f;
        for (int k = 0; k < Hc; ++k) s += x[k] * W[(size_t)n * Hc + k];
        out[(size_t)b * Hc + n] = fmaxf(s + bias[n], 0.f);
    }
}

__global__ void dot_head_kernel(const float* __restrict__ feats,
                                const float* __restrict__ w,
                                const float* __restrict__ bias,
                                float* __restrict__ out) {
    int b = blockIdx.x, t = threadIdx.x;
    float v = 0.f;
#pragma unroll
    for (int i = 0; i < 3; ++i) {
        int c = t + i * 256;
        v += feats[(size_t)b * Hc + c] * w[c];
    }
    float r = blk_sum256(v);
    if (t == 0) out[b] = r + bias[0];
}

// BiLSTM recurrence; pre = x@w_ih^T + b_ih + b_hh precomputed (B,N,1536)
__global__ void lstm_kernel(const float* __restrict__ pre_f,
                            const float* __restrict__ pre_b,
                            const float* __restrict__ whh_f,
                            const float* __restrict__ whh_b, int N,
                            float* __restrict__ out) {
    int b = blockIdx.x, dir = blockIdx.y, j = threadIdx.x;   // j in 0..383
    const float* pre = dir ? pre_b : pre_f;
    const float* whh = dir ? whh_b : whh_f;
    __shared__ float hsh[HDc];
    float c = 0.f;
    hsh[j] = 0.f;
    __syncthreads();
    for (int st = 0; st < N; ++st) {
        int t = dir ? (N - 1 - st) : st;
        const float* p = pre + ((size_t)(b * N + t)) * Gc;
        float gi = p[j], gf = p[HDc + j], gg = p[2 * HDc + j], go = p[3 * HDc + j];
        const float* wi = whh + (size_t)j * HDc;
        const float* wf = whh + (size_t)(HDc + j) * HDc;
        const float* wg = whh + (size_t)(2 * HDc + j) * HDc;
        const float* wo = whh + (size_t)(3 * HDc + j) * HDc;
        for (int k = 0; k < HDc; ++k) {
            float hv = hsh[k];
            gi += hv * wi[k]; gf += hv * wf[k];
            gg += hv * wg[k]; go += hv * wo[k];
        }
        float si = 1.f / (1.f + __expf(-gi));
        float sf = 1.f / (1.f + __expf(-gf));
        float so = 1.f / (1.f + __expf(-go));
        c = sf * c + si * tanhf(gg);
        float hv = so * tanhf(c);
        out[((size_t)(b * N + t)) * Hc + dir * HDc + j] = hv;
        __syncthreads();
        hsh[j] = hv;
        __syncthreads();
    }
}

__global__ void combine_kernel(const float* a, const float* b, const float* c,
                               const float* d, float* out) {
    int i = threadIdx.x;
    if (i < Bc) out[i] = a[i] + b[i] + c[i] + d[i];
}

// ---------------------------------------------------------------------------
// Host orchestration
// ---------------------------------------------------------------------------
extern "C" void kernel_launch(void* const* d_in, const int* in_sizes, int n_in,
                              void* d_out, int out_size, void* d_ws,
                              size_t ws_size, hipStream_t stream) {
    (void)in_sizes; (void)n_in; (void)out_size; (void)ws_size;
    auto F = [&](int i) { return (const float*)d_in[i]; };
    auto I = [&](int i) { return (const int*)d_in[i]; };

    const float* hs = F(0);
    const float* lasth = F(1);
    const float* amask = F(2);
    const int *p_head = I(3), *p_tail = I(4);
    const float* p_mask = F(5);
    const int *s_head = I(6), *s_tail = I(7);
    const float* s_mask = F(8);

    struct AttnRaw { const float *bk,*bo,*bq,*bv,*ln_b,*ln_g,*wk,*wo,*wq,*wv; };
    auto attn_raw = [&](int base) {
        AttnRaw a;
        a.bk = F(base + 0); a.bo = F(base + 1); a.bq = F(base + 2);
        a.bv = F(base + 3); a.ln_b = F(base + 4); a.ln_g = F(base + 5);
        a.wk = F(base + 6); a.wo = F(base + 7); a.wq = F(base + 8);
        a.wv = F(base + 9);
        return a;
    };
    AttnRaw mainA = attn_raw(9), paraA = attn_raw(26), sentA = attn_raw(48);
    const float* ctx_b = F(19);
    const float* ctx_cls_b = F(20);
    const float* ctx_cls_w = F(21);
    const float* ctx_w = F(22);
    const float* layer_w = F(23);
    const float* mha_ln_b = F(24);
    const float* mha_ln_g = F(25);

    // ---- workspace arena ----
    char* wsb = (char*)d_ws;
    size_t off = 0;
    auto alloc = [&](size_t bytes) -> void* {
        void* p = wsb + off;
        off = (off + bytes + 255) & ~(size_t)255;
        return p;
    };
    const size_t WW = (size_t)Hc * Hc;
    const size_t WIH = (size_t)Hc * Gc;
    const size_t MAINE = (size_t)Bc * Lc * Hc;
    const size_t BRE = (size_t)Bc * 64 * Hc;

    auto u16 = [&](size_t n) { return (unsigned short*)alloc(n * 2); };
    auto f32 = [&](size_t n) { return (float*)alloc(n * 4); };

    struct AttnW { unsigned short *wq,*wk,*wv,*wo;
                   const float *bq,*bk,*bv,*bo,*lng,*lnb; };
    auto make_attnw = [&](const AttnRaw& r) {
        AttnW w;
        w.wq = u16(WW); w.wk = u16(WW); w.wv = u16(WW); w.wo = u16(WW);
        w.bq = r.bq; w.bk = r.bk; w.bv = r.bv; w.bo = r.bo;
        w.lng = r.ln_g; w.lnb = r.ln_b;
        return w;
    };
    AttnW mW = make_attnw(mainA), pW = make_attnw(paraA), sW = make_attnw(sentA);
    unsigned short* wt_ih_pf = u16(WIH);
    unsigned short* wt_ih_pb = u16(WIH);
    unsigned short* wt_ih_sf = u16(WIH);
    unsigned short* wt_ih_sb = u16(WIH);

    float* enc0_f = f32(MAINE);
    unsigned short* enc0_b = u16(MAINE);
    unsigned short* Qb = u16(MAINE);
    unsigned short* Kb = u16(MAINE);
    unsigned short* Vb = u16(MAINE);
    unsigned short* Kt = u16(MAINE);
    unsigned short* Cx = u16(MAINE);
    float* Of = f32(MAINE);
    float* enc_f = f32(MAINE);

    float* cls_feats = f32(Bc * Hc);
    float* mean_feats = f32(Bc * Hc);
    float* bfeats = f32(Bc * Hc);
    float* log_ctx = f32(Bc);
    float* log_mean = f32(Bc);
    float* log_para = f32(Bc);
    float* log_sent = f32(Bc);

    float* vraw = f32(BRE);
    float* v_f = f32(BRE);
    unsigned short* v_b = u16(BRE);
    float* bsum_f = f32(Gc);
    float* bsum_b = f32(Gc);
    float* pre_f = f32((size_t)Bc * 64 * Gc);
    float* pre_b = f32((size_t)Bc * 64 * Gc);
    float* lstm_f = f32(BRE);
    unsigned short* lstm_b16 = u16(BRE);
    unsigned short* Qb2 = u16(BRE);
    unsigned short* Kb2 = u16(BRE);
    unsigned short* Vb2 = u16(BRE);
    unsigned short* Kt2 = u16(BRE);
    unsigned short* Cx2 = u16(BRE);
    float* Of2 = f32(BRE);
    float* battn = f32(BRE);

    // ---- launch helpers ----
    auto trw = [&](const float* W, unsigned short* Wt, int N, int K) {
        size_t tot = (size_t)N * K;
        transpose_w_kernel<<<dim3((unsigned)((tot + 255) / 256)), dim3(256), 0,
                             stream>>>(W, Wt, N, K);
    };
    auto gemm = [&](const unsigned short* A, const unsigned short* Bt,
                    const float* bias, int M, int N, int K, float* of,
                    unsigned short* ob) {
        gemm_bf16_kernel<<<dim3(N / 64, M / 64), dim3(128), 0, stream>>>(
            A, Bt, bias, M, N, K, of, ob);
    };
    const float scl = 1.0f / sqrtf((float)Dc);
    auto run_attn = [&](const unsigned short* x_b16, const float* x_res, int Lx,
                        const float* mk, const AttnW& aw, unsigned short* qb,
                        unsigned short* kb, unsigned short* vb,
                        unsigned short* kt, unsigned short* cx, float* of,
                        float* out_f) {
        int M = Bc * Lx;
        gemm(x_b16, aw.wq, aw.bq, M, Hc, Hc, nullptr, qb);
        gemm(x_b16, aw.wk, aw.bk, M, Hc, Hc, nullptr, kb);
        gemm(x_b16, aw.wv, aw.bv, M, Hc, Hc, nullptr, vb);
        size_t tot = (size_t)M * Hc;
        transpose_k_kernel<<<dim3((unsigned)((tot + 255) / 256)), dim3(256), 0,
                             stream>>>(kb, kt, Lx);
        int Lpad = (Lx + 31) & ~31;
        attention_kernel<<<dim3(Lx / 16, NHc, Bc), dim3(32),
                           16 * Lpad * sizeof(float), stream>>>(qb, kt, vb, mk,
                                                                Lx, scl, cx);
        gemm(cx, aw.wo, aw.bo, M, Hc, Hc, of, nullptr);
        add_ln_kernel<<<dim3(M), dim3(256), 0, stream>>>(
            of, x_res, aw.lng, aw.lnb, 1e-12f, out_f, nullptr);
    };

    // ---- 0: weight prep ----
    auto prep_attnw = [&](const AttnRaw& r, const AttnW& w) {
        trw(r.wq, w.wq, Hc, Hc); trw(r.wk, w.wk, Hc, Hc);
        trw(r.wv, w.wv, Hc, Hc); trw(r.wo, w.wo, Hc, Hc);
    };
    prep_attnw(mainA, mW); prep_attnw(paraA, pW); prep_attnw(sentA, sW);
    trw(F(47), wt_ih_pf, Gc, Hc);   // para w_ih_f
    trw(F(46), wt_ih_pb, Gc, Hc);   // para w_ih_b
    trw(F(69), wt_ih_sf, Gc, Hc);   // sent w_ih_f
    trw(F(68), wt_ih_sb, Gc, Hc);   // sent w_ih_b

    // ---- 1: context pooler ----
    gemv_relu_kernel<<<dim3(Bc), dim3(256), 0, stream>>>(lasth, ctx_w, ctx_b,
                                                         cls_feats);
    dot_head_kernel<<<dim3(Bc), dim3(256), 0, stream>>>(cls_feats, ctx_cls_w,
                                                        ctx_cls_b, log_ctx);

    // ---- 2: layer mix + LN (eps 1e-7) ----
    layer_mix_ln_kernel<<<dim3(Bc * Lc), dim3(256), 0, stream>>>(
        hs, layer_w, mha_ln_g, mha_ln_b, enc0_f, enc0_b);

    // ---- 3: main BertAttention ----
    run_attn(enc0_b, enc0_f, Lc, amask, mW, Qb, Kb, Vb, Kt, Cx, Of, enc_f);

    // ---- 4: mean pool + reused classifier ----
    mean_pool_kernel<<<dim3(Bc), dim3(256), 0, stream>>>(enc_f, amask, Lc,
                                                         mean_feats);
    dot_head_kernel<<<dim3(Bc), dim3(256), 0, stream>>>(mean_feats, ctx_cls_w,
                                                        ctx_cls_b, log_mean);

    // ---- 5: branches ----
    auto run_branch = [&](int N, const int* head, const int* tail,
                          const float* mk, const float* ln_g, const float* ln_b,
                          const float* b_ih_f, const float* b_hh_f,
                          const float* b_ih_b, const float* b_hh_b,
                          const unsigned short* wih_f,
                          const unsigned short* wih_b, const float* whh_f,
                          const float* whh_b, const AttnW& aw,
                          const float* cls_w, const float* cls_bias,
                          float* out_logit) {
        int M = Bc * N;
        span_mean_kernel<<<dim3(N, Bc), dim3(256), 0, stream>>>(enc_f, head,
                                                                tail, N, vraw);
        add_ln_kernel<<<dim3(M), dim3(256), 0, stream>>>(vraw, nullptr, ln_g,
                                                         ln_b, 1e-7f, v_f, v_b);
        bias_sum_kernel<<<dim3(6), dim3(256), 0, stream>>>(b_ih_f, b_hh_f,
                                                           bsum_f, Gc);
        bias_sum_kernel<<<dim3(6), dim3(256), 0, stream>>>(b_ih_b, b_hh_b,
                                                           bsum_b, Gc);
        gemm(v_b, wih_f, bsum_f, M, Gc, Hc, pre_f, nullptr);
        gemm(v_b, wih_b, bsum_b, M, Gc, Hc, pre_b, nullptr);
        lstm_kernel<<<dim3(Bc, 2), dim3(HDc), 0, stream>>>(pre_f, pre_b, whh_f,
                                                           whh_b, N, lstm_f);
        size_t tot = (size_t)M * Hc;
        to_bf16_kernel<<<dim3((unsigned)((tot + 255) / 256)), dim3(256), 0,
                         stream>>>(lstm_f, lstm_b16, tot);
        run_attn(lstm_b16, lstm_f, N, mk, aw, Qb2, Kb2, Vb2, Kt2, Cx2, Of2,
                 battn);
        mean_pool_kernel<<<dim3(Bc), dim3(256), 0, stream>>>(battn, mk, N,
                                                             bfeats);
        dot_head_kernel<<<dim3(Bc), dim3(256), 0, stream>>>(bfeats, cls_w,
                                                            cls_bias, out_logit);
    };

    run_branch(16, p_head, p_tail, p_mask, F(39), F(38),
               F(43), F(41), F(42), F(40),
               wt_ih_pf, wt_ih_pb, F(45), F(44),
               pW, F(37), F(36), log_para);
    run_branch(64, s_head, s_tail, s_mask, F(61), F(60),
               F(65), F(63), F(64), F(62),
               wt_ih_sf, wt_ih_sb, F(67), F(66),
               sW, F(59), F(58), log_sent);

    // ---- 6: combine ----
    combine_kernel<<<dim3(1), dim3(32), 0, stream>>>(log_ctx, log_mean,
                                                     log_para, log_sent,
                                                     (float*)d_out);
}
